// KMA_19310172962818
// MI455X (gfx1250) — matrix operand order repaired and checked by reading, not compiled, not run on hardware
//
#include <hip/hip_runtime.h>
#include <hip/hip_bf16.h>

// Problem dims (fixed by the reference)
#define LDIM 4
#define BDIM 4
#define SDIM 2048
#define HDIM 1024
#define HKDIM 1024
#define INTERDIM 4096
#define MDIM (BDIM * SDIM)   // 8192 rows

typedef __attribute__((ext_vector_type(16))) __bf16 v16bf;
typedef __attribute__((ext_vector_type(8)))  float  v8f;

__device__ __forceinline__ unsigned short f2bf(float f) {
  unsigned int u = __float_as_uint(f);
  u += 0x7FFFu + ((u >> 16) & 1u);           // round-to-nearest-even
  return (unsigned short)(u >> 16);
}

// ---------------------------------------------------------------------------
// fp32 -> bf16 raw conversion (grid-stride)
// ---------------------------------------------------------------------------
__global__ void cvt_bf16_kernel(const float* __restrict__ src,
                                unsigned short* __restrict__ dst, long n) {
  long i = (long)blockIdx.x * blockDim.x + threadIdx.x;
  long stride = (long)gridDim.x * blockDim.x;
  for (; i < n; i += stride) dst[i] = f2bf(src[i]);
}

// ---------------------------------------------------------------------------
// Generic bf16 GEMM:  C[z] = A (MxK, row-major) * B[z] ([N,K] row-major) + bias
// Compile-time RELU / SF32 / SB16 -> branch-free epilogue.
// Block = 256 threads = 8 waves; block tile 128x128; wave tile 32x64;
// K-step 32; double-buffered LDS (one barrier per K-step).
// ---------------------------------------------------------------------------
#define LDSROW 40   // padded row stride in halves (80 B -> 16B aligned, conflict-free)

template <bool RELU, bool SF32, bool SB16>
__global__ __launch_bounds__(256)
void gemm_bf16_kernel(const unsigned short* __restrict__ A,
                      const unsigned short* __restrict__ Bbase, long strideB,
                      const float* __restrict__ biasBase, long strideBias,
                      float* __restrict__ Cf, unsigned short* __restrict__ Cb,
                      long strideC, int N, int Kd) {
  const int z = blockIdx.z;
  const unsigned short* __restrict__ B = Bbase + (long)z * strideB;
  const float* __restrict__ bias = biasBase + (long)z * strideBias;

  // double-buffered tiles: [2][128 rows][40 halves]
  __shared__ __align__(16) unsigned short As[2][128 * LDSROW];
  __shared__ __align__(16) unsigned short Bs[2][128 * LDSROW];

  const int t    = threadIdx.x;
  const int w    = t >> 5;
  const int lane = t & 31;
  const int lo   = lane & 15;
  const int hi   = lane >> 4;
  const int wrow = w >> 1;   // 0..3 : which 32-row band
  const int wcol = w & 1;    // 0..1 : which 64-col band

  const int row0 = blockIdx.y * 128;
  const int col0 = blockIdx.x * 128;

  v8f acc[2][4];
#pragma unroll
  for (int mi = 0; mi < 2; ++mi)
#pragma unroll
    for (int ni = 0; ni < 4; ++ni)
      acc[mi][ni] = (v8f){0.f, 0.f, 0.f, 0.f, 0.f, 0.f, 0.f, 0.f};

  const int srow  = t >> 1;          // 0..127 : staged row
  const int spart = (t & 1) * 16;    // 0 or 16 : half-row (16 bf16 = 32B)
  const unsigned short* gA = A + (long)(row0 + srow) * Kd + spart;
  const unsigned short* gB = B + (long)(col0 + srow) * Kd + spart;
  const int sofs = srow * LDSROW + spart;

  const int niter = Kd / 32;

  // ---- prologue: stage k-tile 0 into buffer 0 ----
  {
    uint4 a0 = *(const uint4*)gA;
    uint4 a1 = *(const uint4*)(gA + 8);
    uint4 b0 = *(const uint4*)gB;
    uint4 b1 = *(const uint4*)(gB + 8);
    *(uint4*)&As[0][sofs]     = a0;
    *(uint4*)&As[0][sofs + 8] = a1;
    *(uint4*)&Bs[0][sofs]     = b0;
    *(uint4*)&Bs[0][sofs + 8] = b1;
  }

  for (int it = 0; it < niter; ++it) {
    __syncthreads();                 // LDS[cur] writes visible; prior reads of LDS[nxt] done
    const int cur = it & 1;
    const int nxt = cur ^ 1;
    const bool more = (it + 1) < niter;

    // issue next tile's global loads early (hide latency behind WMMAs)
    uint4 a0, a1, b0, b1;
    if (more) {
      const unsigned short* pa = gA + (long)(it + 1) * 32;
      const unsigned short* pb = gB + (long)(it + 1) * 32;
      a0 = *(const uint4*)pa;  a1 = *(const uint4*)(pa + 8);
      b0 = *(const uint4*)pb;  b1 = *(const uint4*)(pb + 8);
      if ((it + 2) < niter) {   // L2 prefetch two tiles ahead
        __builtin_prefetch(gA + (long)(it + 2) * 32, 0, 0);
        __builtin_prefetch(gB + (long)(it + 2) * 32, 0, 0);
      }
    }

    // A fragment: lanes 0-15 row lo, K {0..7,16..23}; lanes 16-31 K {8..15,24..31}
    v16bf afrag[2], bfrag[4];
#pragma unroll
    for (int mi = 0; mi < 2; ++mi) {
      const int base = (wrow * 32 + mi * 16 + lo) * LDSROW + hi * 8;
      union { uint4 u[2]; v16bf v; } tmp;
      tmp.u[0] = *(const uint4*)&As[cur][base];
      tmp.u[1] = *(const uint4*)&As[cur][base + 16];
      afrag[mi] = tmp.v;
    }
    // B fragment: lanes 0-15 col lo K {0..7,16..23}; lanes 16-31 K {8..15,24..31}
#pragma unroll
    for (int ni = 0; ni < 4; ++ni) {
      const int base = (wcol * 64 + ni * 16 + lo) * LDSROW + hi * 8;
      union { uint4 u[2]; v16bf v; } tmp;
      tmp.u[0] = *(const uint4*)&Bs[cur][base];
      tmp.u[1] = *(const uint4*)&Bs[cur][base + 16];
      bfrag[ni] = tmp.v;
    }

#pragma unroll
    for (int mi = 0; mi < 2; ++mi)
#pragma unroll
      for (int ni = 0; ni < 4; ++ni)
        acc[mi][ni] = __builtin_amdgcn_wmma_f32_16x16x32_bf16(
            false, afrag[mi], false, bfrag[ni], (short)0, acc[mi][ni],
            false, false);

    // park next tile in the alternate buffer (safe: its readers passed the
    // barrier at the top of this iteration)
    if (more) {
      *(uint4*)&As[nxt][sofs]     = a0;
      *(uint4*)&As[nxt][sofs + 8] = a1;
      *(uint4*)&Bs[nxt][sofs]     = b0;
      *(uint4*)&Bs[nxt][sofs + 8] = b1;
    }
  }

  // Epilogue: C/D layout — VGPR r holds M=r (lanes 0-15), M=r+8 (lanes 16-31)
#pragma unroll
  for (int ni = 0; ni < 4; ++ni) {
    const int gc = col0 + wcol * 64 + ni * 16 + lo;
    const float bv = bias[gc];
#pragma unroll
    for (int mi = 0; mi < 2; ++mi) {
      const int gr0 = row0 + wrow * 32 + mi * 16 + hi * 8;
      long idx = (long)z * strideC + (long)gr0 * N + gc;
#pragma unroll
      for (int r = 0; r < 8; ++r, idx += N) {
        float v = acc[mi][ni][r] + bv;
        if (RELU) v = fmaxf(v, 0.f);
        if (SF32) Cf[idx] = v;
        if (SB16) Cb[idx] = f2bf(v);
      }
    }
  }
}

// ---------------------------------------------------------------------------
// Final: energy_inter = <O_l, q_inter>, softmax over L, weighted sum, tanh.
// One 256-thread block per row; each thread owns 4 of the 1024 columns.
// ---------------------------------------------------------------------------
__global__ __launch_bounds__(256)
void finalize_kernel(const float* __restrict__ O,    // [L][M][HK]
                     const float* __restrict__ Qi,   // [M][HK]
                     float* __restrict__ out) {      // [M][HK]
  const int row = blockIdx.x;
  const int t = threadIdx.x;
  const int k = t * 4;

  const float4 q = *(const float4*)&Qi[(long)row * HKDIM + k];
  float4 o[LDIM];
  float part[LDIM];
#pragma unroll
  for (int l = 0; l < LDIM; ++l) {
    o[l] = *(const float4*)&O[((long)l * MDIM + row) * HKDIM + k];
    part[l] = o[l].x * q.x + o[l].y * q.y + o[l].z * q.z + o[l].w * q.w;
  }
  // wave32 reduction
#pragma unroll
  for (int off = 16; off > 0; off >>= 1)
#pragma unroll
    for (int l = 0; l < LDIM; ++l)
      part[l] += __shfl_xor(part[l], off, 32);

  __shared__ float s[8][LDIM];
  if ((t & 31) == 0)
#pragma unroll
    for (int l = 0; l < LDIM; ++l) s[t >> 5][l] = part[l];
  __syncthreads();

  float e[LDIM];
#pragma unroll
  for (int l = 0; l < LDIM; ++l) {
    float a = 0.f;
#pragma unroll
    for (int wv = 0; wv < 8; ++wv) a += s[wv][l];
    e[l] = a;
  }
  float mx = e[0];
#pragma unroll
  for (int l = 1; l < LDIM; ++l) mx = fmaxf(mx, e[l]);
  float wgt[LDIM], wsum = 0.f;
#pragma unroll
  for (int l = 0; l < LDIM; ++l) { wgt[l] = __expf(e[l] - mx); wsum += wgt[l]; }
  const float inv = 1.f / wsum;

  float4 r = make_float4(0.f, 0.f, 0.f, 0.f);
#pragma unroll
  for (int l = 0; l < LDIM; ++l) {
    const float c = wgt[l] * inv;
    r.x += o[l].x * c; r.y += o[l].y * c; r.z += o[l].z * c; r.w += o[l].w * c;
  }
  float4 res;
  res.x = tanhf(r.x); res.y = tanhf(r.y); res.z = tanhf(r.z); res.w = tanhf(r.w);
  *(float4*)&out[(long)row * HKDIM + k] = res;
}

// ---------------------------------------------------------------------------
extern "C" void kernel_launch(void* const* d_in, const int* in_sizes, int n_in,
                              void* d_out, int out_size, void* d_ws, size_t ws_size,
                              hipStream_t stream) {
  const float* embeds = (const float*)d_in[0];   // [B,S,H]
  const float* Wqi    = (const float*)d_in[1];   // [L,HK,H]
  const float* bqi    = (const float*)d_in[2];   // [L,HK]
  const float* Wqx    = (const float*)d_in[3];   // [HK,H]
  const float* bqx    = (const float*)d_in[4];   // [HK]
  const float* Kmem   = (const float*)d_in[5];   // [L,INTER,HK]
  const float* Kb     = (const float*)d_in[6];   // [L,INTER]
  const float* Vmem   = (const float*)d_in[7];   // [L,HK,INTER]
  const float* Vb     = (const float*)d_in[8];   // [L,HK]
  float* out = (float*)d_out;

  // Workspace carve-up (256B aligned)
  char* ws = (char*)d_ws;
  auto take = [&](size_t bytes) -> char* {
    char* p = ws;
    ws += (bytes + 255) & ~(size_t)255;
    return p;
  };
  unsigned short* Eb   = (unsigned short*)take((size_t)MDIM * HDIM * 2);
  unsigned short* Wb   = (unsigned short*)take((size_t)5 * HKDIM * HDIM * 2);
  unsigned short* Kb16 = (unsigned short*)take((size_t)LDIM * INTERDIM * HKDIM * 2);
  unsigned short* Vb16 = (unsigned short*)take((size_t)LDIM * HKDIM * INTERDIM * 2);
  unsigned short* Qb   = (unsigned short*)take((size_t)LDIM * MDIM * HKDIM * 2);
  float*          Qint = (float*)take((size_t)MDIM * HKDIM * 4);
  unsigned short* Abuf = (unsigned short*)take((size_t)MDIM * INTERDIM * 2);
  float*          Obuf = (float*)take((size_t)LDIM * MDIM * HKDIM * 4);

  const dim3 blk(256);
  // --- bf16 conversions ---
  cvt_bf16_kernel<<<2048, 256, 0, stream>>>(embeds, Eb, (long)MDIM * HDIM);
  cvt_bf16_kernel<<<2048, 256, 0, stream>>>(Wqi, Wb, (long)LDIM * HKDIM * HDIM);
  cvt_bf16_kernel<<<1024, 256, 0, stream>>>(Wqx, Wb + (size_t)LDIM * HKDIM * HDIM,
                                            (long)HKDIM * HDIM);
  cvt_bf16_kernel<<<4096, 256, 0, stream>>>(Kmem, Kb16, (long)LDIM * INTERDIM * HKDIM);
  cvt_bf16_kernel<<<4096, 256, 0, stream>>>(Vmem, Vb16, (long)LDIM * HKDIM * INTERDIM);

  // --- GEMM1: per-layer inner queries (bf16 out), z batches the 4 layers ---
  gemm_bf16_kernel<false, false, true>
      <<<dim3(HKDIM / 128, MDIM / 128, LDIM), blk, 0, stream>>>(
      Eb, Wb, (long)HKDIM * HDIM, bqi, HKDIM,
      nullptr, Qb, (long)MDIM * HKDIM, HKDIM, HDIM);

  // --- GEMM1b: inter query (fp32 out) ---
  gemm_bf16_kernel<false, true, false>
      <<<dim3(HKDIM / 128, MDIM / 128, 1), blk, 0, stream>>>(
      Eb, Wb + (size_t)LDIM * HKDIM * HDIM, 0, bqx, 0,
      Qint, nullptr, 0, HKDIM, HDIM);

  // --- per layer: energy+relu (GEMM2) then value readout (GEMM3) ---
  for (int l = 0; l < LDIM; ++l) {
    gemm_bf16_kernel<true, false, true>
        <<<dim3(INTERDIM / 128, MDIM / 128, 1), blk, 0, stream>>>(
        Qb + (size_t)l * MDIM * HKDIM,
        Kb16 + (size_t)l * INTERDIM * HKDIM, 0,
        Kb + (size_t)l * INTERDIM, 0,
        nullptr, Abuf, 0, INTERDIM, HKDIM);

    gemm_bf16_kernel<false, true, false>
        <<<dim3(HKDIM / 128, MDIM / 128, 1), blk, 0, stream>>>(
        Abuf,
        Vb16 + (size_t)l * HKDIM * INTERDIM, 0,
        Vb + (size_t)l * HKDIM, 0,
        Obuf + (size_t)l * MDIM * HKDIM, nullptr, 0, HKDIM, INTERDIM);
  }

  // --- softmax over layers + tanh ---
  finalize_kernel<<<MDIM, 256, 0, stream>>>(Obuf, Qint, out);
}
